// MultiHeadAttention_57930518888563
// MI455X (gfx1250) — matrix operand order repaired and checked by reading, not compiled, not run on hardware
//
#include <hip/hip_runtime.h>
#include <stdint.h>

// Problem constants (reference: B=2, S=2048, D=1024, H=16, dk=64, TEMP=1.5)
#define Hn 16
#define Dm 1024
#define DK 64
#define Sq 2048
#define Bb 2
#define Tt (Bb * Sq)  // 4096 tokens

typedef __attribute__((ext_vector_type(16))) __bf16 v16bf;
typedef __attribute__((ext_vector_type(8)))  float  v8f;

union FragB {
    v16bf v;
    uint4 q[2];
};

__device__ __forceinline__ uint16_t f2bf(float f) {
    union { float f; uint32_t u; } c;
    c.f = f;
    uint32_t u = c.u;
    uint32_t r = (u + 0x7FFFu + ((u >> 16) & 1u)) >> 16;  // RNE
    return (uint16_t)r;
}

// ---------------------------------------------------------------- convert
__global__ __launch_bounds__(256) void k_f32_to_bf16(const float* __restrict__ src,
                                                     uint16_t* __restrict__ dst, int n) {
    int i = blockIdx.x * blockDim.x + threadIdx.x;
    int stride = gridDim.x * blockDim.x;
    for (; i < n; i += stride) dst[i] = f2bf(src[i]);
}

// ---------------------------------------------------------------- projection GEMM
// Y = X @ W^T + bias.  X: Tt x Dm bf16 row-major. W: Dm x Dm bf16 row-major (out,in).
// One wave computes a 32x64 output tile: 2 M-tiles x 4 N-tiles of 16x16 WMMAs.
// Per K=32 step: 4 A-frag loads + 8 B-frag loads feed 8 v_wmma_f32_16x16x32_bf16.
// mode 0: bf16 store, (b,h,s,d) layout (Q / K)
// mode 1: bf16 store, (b,h,d,s) layout (V transposed for P@V B-operand)
// mode 2: f32 store, row-major (token, feature) -> final output
__global__ __launch_bounds__(32) void k_proj(const uint16_t* __restrict__ X,
                                             const uint16_t* __restrict__ W,
                                             const float* __restrict__ bias,
                                             uint16_t* __restrict__ out_bf,
                                             float* __restrict__ out_f,
                                             int mode) {
    const int lane = threadIdx.x;
    const int lm = lane & 15, half = lane >> 4;
    const int M0 = blockIdx.x * 32, N0 = blockIdx.y * 64;

    const uint16_t* arow[2];
    const uint16_t* brow[4];
#pragma unroll
    for (int mt = 0; mt < 2; ++mt)
        arow[mt] = X + (size_t)(M0 + mt * 16 + lm) * Dm;            // A: row per lane
#pragma unroll
    for (int nt = 0; nt < 4; ++nt)
        brow[nt] = W + (size_t)(N0 + nt * 16 + lm) * Dm + half * 16; // B: col per lane

    v8f acc[2][4] = {};

#pragma unroll 2
    for (int k0 = 0; k0 < Dm; k0 += 32) {
        FragB a[2], b[4];
#pragma unroll
        for (int mt = 0; mt < 2; ++mt) {
            a[mt].q[0] = *(const uint4*)(arow[mt] + k0 + half * 16);       // K = half*16..+7
            a[mt].q[1] = *(const uint4*)(arow[mt] + k0 + half * 16 + 8);  // K = half*16+8..+15
        }
#pragma unroll
        for (int nt = 0; nt < 4; ++nt) {
            b[nt].q[0] = *(const uint4*)(brow[nt] + k0);      // K = half*16..+7
            b[nt].q[1] = *(const uint4*)(brow[nt] + k0 + 8);  // K = half*16+8..+15
        }
        if (k0 + 128 < Dm) {  // pull next tiles toward the WGP (global_prefetch_b8)
            __builtin_prefetch(arow[0] + k0 + 128, 0, 3);
            __builtin_prefetch(arow[1] + k0 + 128, 0, 3);
        }
#pragma unroll
        for (int mt = 0; mt < 2; ++mt)
#pragma unroll
            for (int nt = 0; nt < 4; ++nt)
                acc[mt][nt] = __builtin_amdgcn_wmma_f32_16x16x32_bf16(
                    false, a[mt].v, false, b[nt].v, (short)0, acc[mt][nt], false, false);
    }

#pragma unroll
    for (int nt = 0; nt < 4; ++nt) {
        const int n = N0 + nt * 16 + lm;  // output feature (column) for this lane
        const float bn = bias[n];
        const int h = n >> 6, d = n & 63;
#pragma unroll
        for (int mt = 0; mt < 2; ++mt) {
            if (mode == 2) {
#pragma unroll
                for (int i = 0; i < 8; ++i) {
                    int m = M0 + mt * 16 + half * 8 + i;  // C: lanes 0-15 rows 0..7, 16-31 rows 8..15
                    out_f[(size_t)m * Dm + n] = acc[mt][nt][i] + bn;
                }
            } else {
#pragma unroll
                for (int i = 0; i < 8; ++i) {
                    int m = M0 + mt * 16 + half * 8 + i;  // token index
                    int bi = m >> 11, s = m & (Sq - 1);   // batch, seq pos
                    size_t idx;
                    if (mode == 0)
                        idx = (((size_t)(bi * Hn + h)) * Sq + s) * DK + d;
                    else
                        idx = (((size_t)(bi * Hn + h)) * DK + d) * Sq + s;
                    out_bf[idx] = f2bf(acc[mt][nt][i] + bn);
                }
            }
        }
    }
}

// ---------------------------------------------------------------- flash attention
// One wave per (b, h, 16-query tile); online softmax over 32-key tiles.
__global__ __launch_bounds__(32) void k_flash(const uint16_t* __restrict__ Q,
                                              const uint16_t* __restrict__ K,
                                              const uint16_t* __restrict__ Vt,
                                              uint16_t* __restrict__ AO) {
    __shared__ uint16_t pls[16 * 32];  // P tile: 16 queries x 32 keys, bf16, row-major

    const int lane = threadIdx.x;
    const int lm = lane & 15, half = lane >> 4;
    const int q0 = blockIdx.x * 16;
    const int bh = blockIdx.y;
    const int h = bh & (Hn - 1), b = bh >> 4;

    const uint16_t* Qh = Q + (size_t)bh * Sq * DK;
    const uint16_t* Kh = K + (size_t)bh * Sq * DK;
    const uint16_t* Vh = Vt + (size_t)bh * DK * Sq;

    // Q A-fragments for the two dk k-steps (dk=64 = 2 x K32), kept in registers.
    FragB qa[2];
    {
        const uint16_t* qrow = Qh + (size_t)(q0 + lm) * DK;
#pragma unroll
        for (int s = 0; s < 2; ++s) {
            qa[s].q[0] = *(const uint4*)(qrow + s * 32 + half * 16);
            qa[s].q[1] = *(const uint4*)(qrow + s * 32 + half * 16 + 8);
        }
    }

    const float scale = 1.0f / (8.0f * 1.5f);  // 1/(sqrt(dk)*TEMP) = 1/12

    float mrun[8], lsum[8];
    v8f acc[4] = {};  // 16 queries x 64 dims
#pragma unroll
    for (int i = 0; i < 8; ++i) { mrun[i] = -1e30f; lsum[i] = 0.0f; }

    for (int kt = 0; kt < Sq; kt += 32) {
        // prefetch next key tile's K rows while we do softmax VALU work on this one
        if (kt + 32 < Sq) {
            __builtin_prefetch(Kh + (size_t)(kt + 32 + lm) * DK, 0, 3);
            __builtin_prefetch(Kh + (size_t)(kt + 48 + lm) * DK, 0, 3);
        }

        // ---- scores: two 16-key halves, each Q(16x64) @ K^T(64x16)
        v8f sc[2];
#pragma unroll
        for (int j = 0; j < 2; ++j) {
            v8f c = {};
            const uint16_t* krow = Kh + (size_t)(kt + j * 16 + lm) * DK + half * 16;
#pragma unroll
            for (int s = 0; s < 2; ++s) {
                FragB kb;
                kb.q[0] = *(const uint4*)(krow + s * 32);
                kb.q[1] = *(const uint4*)(krow + s * 32 + 8);
                c = __builtin_amdgcn_wmma_f32_16x16x32_bf16(false, qa[s].v, false, kb.v,
                                                            (short)0, c, false, false);
            }
#pragma unroll
            for (int i = 0; i < 8; ++i) c[i] *= scale;
            sc[j] = c;
        }

        // ---- online softmax (rows split: lanes 0-15 rows 0..7, lanes 16-31 rows 8..15;
        //      columns = keys striped across the 16-lane half-groups)
#pragma unroll
        for (int i = 0; i < 8; ++i) {
            float mt = fmaxf(sc[0][i], sc[1][i]);
#pragma unroll
            for (int msk = 8; msk >= 1; msk >>= 1) mt = fmaxf(mt, __shfl_xor(mt, msk));
            float mnew = fmaxf(mrun[i], mt);
            float corr = __expf(mrun[i] - mnew);
            float p0 = __expf(sc[0][i] - mnew);
            float p1 = __expf(sc[1][i] - mnew);
            float rs = p0 + p1;
#pragma unroll
            for (int msk = 8; msk >= 1; msk >>= 1) rs += __shfl_xor(rs, msk);
            lsum[i] = lsum[i] * corr + rs;
            mrun[i] = mnew;
#pragma unroll
            for (int n4 = 0; n4 < 4; ++n4) acc[n4][i] *= corr;

            // C-layout -> row-major LDS (converts to A-layout for the P@V WMMA)
            int row = half * 8 + i;
            pls[row * 32 + lm]      = f2bf(p0);
            pls[row * 32 + 16 + lm] = f2bf(p1);
        }
        asm volatile("s_wait_dscnt 0" ::: "memory");  // single wave: stores visible

        FragB pa;
        pa.q[0] = *(const uint4*)(&pls[lm * 32 + half * 16]);
        pa.q[1] = *(const uint4*)(&pls[lm * 32 + half * 16 + 8]);

        // ---- out += P(16x32) @ V(32x64): 4 N-tiles of 16 dims
#pragma unroll
        for (int n4 = 0; n4 < 4; ++n4) {
            FragB vb;
            const uint16_t* vrow = Vh + (size_t)(n4 * 16 + lm) * Sq + kt + half * 16;
            vb.q[0] = *(const uint4*)(vrow);
            vb.q[1] = *(const uint4*)(vrow + 8);
            acc[n4] = __builtin_amdgcn_wmma_f32_16x16x32_bf16(false, pa.v, false, vb.v,
                                                              (short)0, acc[n4], false, false);
        }
    }

    // ---- normalize and store to (B,S,D) bf16 for the output projection
#pragma unroll
    for (int i = 0; i < 8; ++i) {
        float inv = 1.0f / lsum[i];
        int srow = q0 + half * 8 + i;
        size_t base = ((size_t)(b * Sq + srow)) * Dm + h * DK;
#pragma unroll
        for (int n4 = 0; n4 < 4; ++n4)
            AO[base + n4 * 16 + lm] = f2bf(acc[n4][i] * inv);
    }
}

// ---------------------------------------------------------------- launch
extern "C" void kernel_launch(void* const* d_in, const int* in_sizes, int n_in,
                              void* d_out, int out_size, void* d_ws, size_t ws_size,
                              hipStream_t stream) {
    const float* x  = (const float*)d_in[0];
    const float* Wq = (const float*)d_in[1];
    const float* bq = (const float*)d_in[2];
    const float* Wk = (const float*)d_in[3];
    const float* bk = (const float*)d_in[4];
    const float* Wv = (const float*)d_in[5];
    const float* bv = (const float*)d_in[6];
    const float* Wo = (const float*)d_in[7];
    const float* bo = (const float*)d_in[8];
    float* out = (float*)d_out;

    const size_t TD = (size_t)Tt * Dm;  // 4M elements
    const size_t DD = (size_t)Dm * Dm;  // 1M elements

    uint16_t* ws  = (uint16_t*)d_ws;    // total 24M u16 = 48 MB
    uint16_t* xb  = ws;
    uint16_t* wqb = xb + TD;
    uint16_t* wkb = wqb + DD;
    uint16_t* wvb = wkb + DD;
    uint16_t* wob = wvb + DD;
    uint16_t* Qb  = wob + DD;   // (B,H,S,dk)
    uint16_t* Kb  = Qb + TD;    // (B,H,S,dk)
    uint16_t* Vtb = Kb + TD;    // (B,H,dk,S)
    uint16_t* aob = Vtb + TD;   // (B,S,D)

    k_f32_to_bf16<<<1024, 256, 0, stream>>>(x,  xb,  (int)TD);
    k_f32_to_bf16<<<512,  256, 0, stream>>>(Wq, wqb, (int)DD);
    k_f32_to_bf16<<<512,  256, 0, stream>>>(Wk, wkb, (int)DD);
    k_f32_to_bf16<<<512,  256, 0, stream>>>(Wv, wvb, (int)DD);
    k_f32_to_bf16<<<512,  256, 0, stream>>>(Wo, wob, (int)DD);

    dim3 gp(Tt / 32, Dm / 64);
    k_proj<<<gp, 32, 0, stream>>>(xb, wqb, bq, Qb,  nullptr, 0);
    k_proj<<<gp, 32, 0, stream>>>(xb, wkb, bk, Kb,  nullptr, 0);
    k_proj<<<gp, 32, 0, stream>>>(xb, wvb, bv, Vtb, nullptr, 1);

    k_flash<<<dim3(Sq / 16, Bb * Hn), 32, 0, stream>>>(Qb, Kb, Vtb, aob);

    k_proj<<<gp, 32, 0, stream>>>(aob, wob, bo, nullptr, out, 2);
}
